// Generator_7894149890558
// MI455X (gfx1250) — compile-verified, matrix operand order checked by reading, not run on hardware
//
#include <hip/hip_runtime.h>
#include <hip/hip_bf16.h>
#include <math.h>

typedef __attribute__((ext_vector_type(16))) _Float16 v16h;
typedef __attribute__((ext_vector_type(8)))  float    v8f;
typedef unsigned int u32x4 __attribute__((ext_vector_type(4)));
typedef int          i32x4 __attribute__((ext_vector_type(4)));
typedef int          i32x8 __attribute__((ext_vector_type(8)));

// ---------------------------------------------------------------------------
// Implicit-GEMM conv using v_wmma_f32_16x16x32_f16.
// Block = 4 waves (128 threads). Block tile = 64 couts x 64 pixels:
//   wave w owns couts [blkCout + 16w, +16), and computes 4 pixel sub-tiles,
//   so the A (weight) tile is reused 4x per wave and the B (im2col) tile is
//   reused by all 4 waves. 4 v_wmma per wave per 32-wide K chunk.
// Flags: relu_in (pre-activation), res (residual add), relu_out.
// ---------------------------------------------------------------------------
__global__ __launch_bounds__(128) void conv_wmma_kernel(
    const float* __restrict__ x, const float* __restrict__ w,
    const float* __restrict__ res, float* __restrict__ y,
    int Cin, int H, int W, int Cout, int Hout, int Wout,
    int KH, int KW, int stride, int pad, int relu_in, int relu_out)
{
  __shared__ _Float16 As[64][36];
  __shared__ _Float16 Bs[64][36];
  const int tid  = threadIdx.x;
  const int lane = tid & 31;
  const int wave = tid >> 5;
  const int pixBase = blockIdx.x * 64;
  const int coutBlk = blockIdx.y * 64;
  const int n   = blockIdx.z;
  const int P   = Hout * Wout;
  const int KHW = KH * KW;
  const int K   = Cin * KHW;
  const int col   = tid & 31;   // k within chunk
  const int rbase = tid >> 5;   // staging row phase (0..3)

  v8f acc[4] = {};

  for (int kc = 0; kc < K; kc += 32) {
    const int kg = kc + col;
    const bool kok = (kg < K);
    int ci = 0, kh = 0, kw = 0;
    if (kok) { ci = kg / KHW; int rr = kg - ci * KHW; kh = rr / KW; kw = rr - kh * KW; }
    const float* xin = x + ((size_t)n * Cin + ci) * (size_t)(H * W);
    if (kg + 32 < K) __builtin_prefetch(&w[(size_t)coutBlk * K + kg + 32], 0, 1);
#pragma unroll
    for (int t = 0; t < 16; ++t) {
      const int r = rbase + 4 * t;               // 0..63
      float av = 0.f;
      const int co = coutBlk + r;
      if (kok && co < Cout) av = w[(size_t)co * K + kg];
      As[r][col] = (_Float16)av;
      float bv = 0.f;
      if (kok) {
        const int p  = pixBase + r;
        const int oh = p / Wout, ow = p - oh * Wout;
        const int ih = oh * stride - pad + kh;
        const int iw = ow * stride - pad + kw;
        if ((unsigned)ih < (unsigned)H && (unsigned)iw < (unsigned)W) {
          bv = xin[ih * W + iw];
          if (relu_in) bv = bv > 0.f ? bv : 0.f;
        }
      }
      Bs[r][col] = (_Float16)bv;
    }
    __syncthreads();
    const int m = lane & 15, half = lane >> 4;
    v16h a = {};
#pragma unroll
    for (int j = 0; j < 8; ++j) {
      const int kb = (j < 4) ? (2 * j + 8 * half) : (16 + 2 * (j - 4) + 8 * half);
      a[2 * j]     = As[wave * 16 + m][kb];
      a[2 * j + 1] = As[wave * 16 + m][kb + 1];
    }
#pragma unroll
    for (int sub = 0; sub < 4; ++sub) {
      v16h b = {};
#pragma unroll
      for (int j = 0; j < 8; ++j) {
        const int kb = (j < 4) ? (2 * j + 8 * half) : (16 + 2 * (j - 4) + 8 * half);
        b[2 * j]     = Bs[sub * 16 + m][kb];
        b[2 * j + 1] = Bs[sub * 16 + m][kb + 1];
      }
      acc[sub] = __builtin_amdgcn_wmma_f32_16x16x32_f16(false, a, false, b, (short)0,
                                                        acc[sub], false, false);
    }
    __syncthreads();
  }

  const int m = lane & 15, half = lane >> 4;
#pragma unroll
  for (int sub = 0; sub < 4; ++sub) {
    const int p = pixBase + sub * 16 + m;
#pragma unroll
    for (int r = 0; r < 8; ++r) {
      const int co = coutBlk + wave * 16 + r + 8 * half;
      if (co < Cout) {
        const size_t oi = ((size_t)n * Cout + co) * (size_t)P + p;
        float v = acc[sub][r];
        if (res) v += res[oi];
        if (relu_out) v = v > 0.f ? v : 0.f;
        y[oi] = v;
      }
    }
  }
}

// ---------------------------------------------------------------------------
// conv_transpose, stride 2, kernel 4x4, SAME (lhs_dilation=2, pad=2, un-flipped
// kernel, per jax.lax.conv_transpose). Output pixels grouped by parity class so
// each tile sees a fixed 2x2 sub-kernel -> dense WMMA GEMM with K = Cin*4.
// Tile = 16 couts x 16 same-parity pixels of one output row (one wave).
// ---------------------------------------------------------------------------
__global__ __launch_bounds__(32) void convt_wmma_kernel(
    const float* __restrict__ x, const float* __restrict__ w,
    float* __restrict__ y, int Cin, int H, int W, int Cout, int relu_out)
{
  __shared__ _Float16 As[16][36];
  __shared__ _Float16 Bs[16][36];
  const int lane  = threadIdx.x;
  const int Wout  = 2 * W;
  const int tilesPerRow = 2 * (W / 16);
  const int tx    = blockIdx.x;
  const int prow  = tx / tilesPerRow;
  const int rem   = tx - prow * tilesPerRow;
  const int qpar  = rem & 1;
  const int qblk  = rem >> 1;
  const int coutBase = blockIdx.y * 16;
  const int n     = blockIdx.z;
  const int ppar  = prow & 1;
  const int K     = Cin * 4;

  v8f acc = {};

  for (int kc = 0; kc < K; kc += 32) {
    const int kg  = kc + lane;
    const bool kok = (kg < K);
    const int ci  = kg >> 2, tap = kg & 3;
    const int kh  = ppar + 2 * (tap >> 1);
    const int kw  = qpar + 2 * (tap & 1);
    const int ih  = (prow + kh - 2) >> 1;   // parity guarantees evenness
    const float* xin = x + ((size_t)n * Cin + ci) * (size_t)(H * W);
#pragma unroll
    for (int t = 0; t < 16; ++t) {
      float av = 0.f;
      const int co = coutBase + t;
      if (kok && co < Cout) av = w[(((size_t)co * Cin + ci) << 4) + kh * 4 + kw];
      As[t][lane] = (_Float16)av;
      float bv = 0.f;
      const int q  = qpar + 2 * (qblk * 16 + t);
      const int iw = (q + kw - 2) >> 1;
      if (kok && (unsigned)ih < (unsigned)H && (unsigned)iw < (unsigned)W)
        bv = xin[ih * W + iw];
      Bs[t][lane] = (_Float16)bv;
    }
    __syncthreads();
    const int m = lane & 15, half = lane >> 4;
    v16h a = {}, b = {};
#pragma unroll
    for (int j = 0; j < 8; ++j) {
      const int kb = (j < 4) ? (2 * j + 8 * half) : (16 + 2 * (j - 4) + 8 * half);
      a[2 * j]     = As[m][kb];
      a[2 * j + 1] = As[m][kb + 1];
      b[2 * j]     = Bs[m][kb];
      b[2 * j + 1] = Bs[m][kb + 1];
    }
    acc = __builtin_amdgcn_wmma_f32_16x16x32_f16(false, a, false, b, (short)0, acc,
                                                 false, false);
    __syncthreads();
  }

  const int pn = lane & 15, half = lane >> 4;
  const int q  = qpar + 2 * (qblk * 16 + pn);
#pragma unroll
  for (int r = 0; r < 8; ++r) {
    const int co = coutBase + r + 8 * half;
    if (co < Cout) {
      const size_t oi = ((size_t)n * Cout + co) * (size_t)(2 * H * Wout) + prow * Wout + q;
      float v = acc[r];
      if (relu_out) v = v > 0.f ? v : 0.f;
      y[oi] = v;
    }
  }
}

// ---------------------------------------------------------------------------
// k = texth @ wk, v = texth @ wv   (thread per (b,t,e); tiny GEMMs)
// ---------------------------------------------------------------------------
__global__ void attn_kv_kernel(const float* __restrict__ texth,
                               const float* __restrict__ wk,
                               const float* __restrict__ wv,
                               float* __restrict__ kbuf, float* __restrict__ vbuf)
{
  const int i = blockIdx.x * blockDim.x + threadIdx.x;   // 32*64*64
  if (i >= 32 * 64 * 64) return;
  const int e = i & 63, t = (i >> 6) & 63, b = i >> 12;
  const float* th = texth + ((size_t)b * 64 + t) * 256;
  float sk = 0.f, sv = 0.f;
  for (int c = 0; c < 256; ++c) {
    const float xv = th[c];
    sk += xv * wk[c * 64 + e];
    sv += xv * wv[c * 64 + e];
  }
  kbuf[i] = sk;
  vbuf[i] = sv;
}

// ---------------------------------------------------------------------------
// Fused cross-attention per token (wave per token; lanes split 64-wide dims).
// ---------------------------------------------------------------------------
__global__ __launch_bounds__(128) void attn_fused_kernel(
    const float* __restrict__ s, const float* __restrict__ kbuf,
    const float* __restrict__ vbuf, const float* __restrict__ wq,
    const float* __restrict__ wo, const unsigned char* __restrict__ mask,
    float* __restrict__ sr)
{
  __shared__ float lds[4][4][64];
  const int wave = threadIdx.x >> 5;
  const int lane = threadIdx.x & 31;
  const int token = blockIdx.x * 4 + wave;   // 131072 tokens exactly
  const int b  = token >> 12;
  const int nn = token & 4095;
  float* tok = lds[wave][0];
  float* qv  = lds[wave][1];
  float* pv  = lds[wave][2];
  float* cx  = lds[wave][3];

  const float* sb = s + (size_t)b * 64 * 4096 + nn;
  tok[lane]      = sb[(size_t)lane * 4096];
  tok[lane + 32] = sb[(size_t)(lane + 32) * 4096];
  __syncthreads();

  for (int rep = 0; rep < 2; ++rep) {
    const int e = lane + 32 * rep;
    float a = 0.f;
    for (int d = 0; d < 64; ++d) a += tok[d] * wq[d * 64 + e];
    qv[e] = a;
  }
  __syncthreads();

  const float* kb = kbuf + (size_t)b * 64 * 64;
  const float* vb = vbuf + (size_t)b * 64 * 64;
  float sc[2];
  for (int rep = 0; rep < 2; ++rep) {
    const int t = lane + 32 * rep;
    float a = 0.f;
    const float* kt = kb + t * 64;
    for (int e = 0; e < 64; ++e) a += qv[e] * kt[e];
    a *= 0.125f;                         // 1/sqrt(64)
    if (!mask[(size_t)b * 64 + t]) a = -1e9f;
    sc[rep] = a;
  }
  float mx = fmaxf(sc[0], sc[1]);
  for (int o = 16; o > 0; o >>= 1) mx = fmaxf(mx, __shfl_xor(mx, o, 32));
  const float e0 = expf(sc[0] - mx), e1 = expf(sc[1] - mx);
  float ssum = e0 + e1;
  for (int o = 16; o > 0; o >>= 1) ssum += __shfl_xor(ssum, o, 32);
  const float inv = 1.f / ssum;
  pv[lane]      = e0 * inv;
  pv[lane + 32] = e1 * inv;
  __syncthreads();

  for (int rep = 0; rep < 2; ++rep) {
    const int e = lane + 32 * rep;
    float a = 0.f;
    for (int t = 0; t < 64; ++t) a += pv[t] * vb[t * 64 + e];
    cx[e] = a;
  }
  __syncthreads();

  float* srb = sr + (size_t)b * 64 * 4096 + nn;
  for (int rep = 0; rep < 2; ++rep) {
    const int e2 = lane + 32 * rep;
    float a = tok[e2];
    for (int e = 0; e < 64; ++e) a += cx[e] * wo[e * 64 + e2];
    srb[(size_t)e2 * 4096] = a;
  }
}

// ---------------------------------------------------------------------------
// ||E_c||^2 for all 512 codes
// ---------------------------------------------------------------------------
__global__ void codebook_sq_kernel(const float* __restrict__ cb, float* __restrict__ esq)
{
  const int c = blockIdx.x * blockDim.x + threadIdx.x;
  if (c >= 512) return;
  float s = 0.f;
  for (int d = 0; d < 64; ++d) { const float v = cb[c * 64 + d]; s += v * v; }
  esq[c] = s;
}

__global__ void zero_counts_kernel(unsigned int* __restrict__ counts)
{
  const int c = blockIdx.x * blockDim.x + threadIdx.x;
  if (c < 512) counts[c] = 0u;
}

// ---------------------------------------------------------------------------
// VQ nearest code via WMMA (-2*F.E^T), fused running argmin over 32 code
// tiles. Code tiles (16 codes x 64 f32) are DMA'd into LDS by the Tensor Data
// Mover (TENSOR_LOAD_TO_LDS), synchronized with s_wait_tensorcnt.
// D# layout per CDNA5 ISA:
//   group0: [1:0]=count=1, [63:32]=lds_addr, [120:64]=global byte addr,
//           [127:126]=type=2
//   group1: [17:16]=data_size=2 (4B), dim0=64, dim1=512, tile0=64, tile1=16,
//           dim0_stride=64
// ---------------------------------------------------------------------------
__global__ __launch_bounds__(32) void vq_argmin_kernel(
    const float* __restrict__ sr, const float* __restrict__ cb,
    const float* __restrict__ esq, int* __restrict__ idx,
    unsigned int* __restrict__ counts)
{
  __shared__ _Float16 Fs[16][68];
  __shared__ float    Esf[16][64];
  const int lane = threadIdx.x;
  const int rowBase = blockIdx.x * 16;

  // stage 16 rows x 64 dims of features (NCHW -> [row, d] gather), f32 -> f16
#pragma unroll
  for (int t = 0; t < 16; ++t) {
    const int row = rowBase + t;
    const int b = row >> 12, pix = row & 4095;
    const float* base = sr + (size_t)b * 262144 + pix;
    Fs[t][lane]      = (_Float16)base[(size_t)lane * 4096];
    Fs[t][lane + 32] = (_Float16)base[(size_t)(lane + 32) * 4096];
  }
  __syncthreads();

  const unsigned ldsA = (unsigned)(size_t)(void*)&Esf[0][0];  // LDS byte offset
  const i32x8 g1 = { 0x00020000,          // data_size = 4B
                     0x00400000,          // tensor_dim0 = 64 (low16 in [31:16])
                     0x02000000,          // tensor_dim1 = 512 (low16 in [31:16])
                     0x00400000,          // tile_dim0 = 64 in [31:16]
                     16,                  // tile_dim1 = 16
                     64,                  // tensor_dim0_stride = 64
                     0, 0 };
  const i32x4 gz = { 0, 0, 0, 0 };
#if defined(__clang_major__) && (__clang_major__ >= 23)
  const i32x8 gz8 = { 0, 0, 0, 0, 0, 0, 0, 0 };
#endif

  const int m = lane & 15, half = lane >> 4;
  float best = 3.4e38f;
  int bidx = 0;

  for (int ct = 0; ct < 32; ++ct) {
    const int cBase = ct * 16;
    const unsigned long long ga =
        (unsigned long long)(size_t)(cb + (size_t)cBase * 64);
    const u32x4 g0 = { 1u, ldsA, (unsigned)(ga & 0xffffffffu),
                       (unsigned)(ga >> 32) | 0x80000000u /* type=2 */ };
#if defined(__clang_major__) && (__clang_major__ >= 23)
    __builtin_amdgcn_tensor_load_to_lds(g0, g1, gz, gz, gz8, 0);
#else
    __builtin_amdgcn_tensor_load_to_lds(g0, g1, gz, gz, 0);
#endif
    __builtin_amdgcn_s_wait_tensorcnt(0);
    __syncthreads();

    v8f acc = {};
#pragma unroll
    for (int c0 = 0; c0 < 64; c0 += 32) {
      v16h a = {}, b = {};
#pragma unroll
      for (int j = 0; j < 8; ++j) {
        const int kb = c0 + ((j < 4) ? (2 * j + 8 * half) : (16 + 2 * (j - 4) + 8 * half));
        a[2 * j]     = (_Float16)Esf[m][kb];
        a[2 * j + 1] = (_Float16)Esf[m][kb + 1];
        b[2 * j]     = Fs[m][kb];
        b[2 * j + 1] = Fs[m][kb + 1];
      }
      acc = __builtin_amdgcn_wmma_f32_16x16x32_f16(false, a, false, b, (short)0, acc,
                                                   false, false);
    }
#pragma unroll
    for (int r = 0; r < 8; ++r) {
      const int code = cBase + r + 8 * half;
      const float d2 = esq[code] - 2.f * acc[r];
      if (d2 < best) { best = d2; bidx = code; }
    }
    __syncthreads();   // LDS reads done before next TDM overwrites Esf
  }

  const float ob = __shfl_xor(best, 16, 32);
  const int   oi = __shfl_xor(bidx, 16, 32);
  if (ob < best || (ob == best && oi < bidx)) { best = ob; bidx = oi; }
  if (lane < 16) {
    idx[rowBase + lane] = bidx;
    atomicAdd(&counts[bidx], 1u);
  }
}

// quantized (forward value of straight-through) = E[idx], scattered to NCHW
__global__ void vq_gather_kernel(const int* __restrict__ idx,
                                 const float* __restrict__ cb,
                                 float* __restrict__ quant)
{
  const int i = blockIdx.x * 256 + threadIdx.x;   // 32*64*4096
  const int pix = i & 4095, d = (i >> 12) & 63, b = i >> 18;
  const int code = idx[b * 4096 + pix];
  quant[i] = cb[code * 64 + d];
}

__global__ void perplexity_kernel(const unsigned int* __restrict__ counts,
                                  float* __restrict__ outp)
{
  __shared__ float red[256];
  const int t = threadIdx.x;
  float local = 0.f;
  for (int c = t; c < 512; c += 256) {
    const float p = (float)counts[c] / 131072.0f;
    local += p * logf(p + 1e-10f);
  }
  red[t] = local;
  __syncthreads();
  for (int s = 128; s > 0; s >>= 1) {
    if (t < s) red[t] += red[t + s];
    __syncthreads();
  }
  if (t == 0) outp[0] = expf(-red[0]);
}

// ---------------------------------------------------------------------------
// Host orchestration
// ---------------------------------------------------------------------------
extern "C" void kernel_launch(void* const* d_in, const int* in_sizes, int n_in,
                              void* d_out, int out_size, void* d_ws, size_t ws_size,
                              hipStream_t stream)
{
  const float* imgh   = (const float*)d_in[0];
  const float* texth  = (const float*)d_in[1];
  const unsigned char* mask = (const unsigned char*)d_in[2];
  const float* enc_w1 = (const float*)d_in[3];
  const float* enc_w2 = (const float*)d_in[4];
  const float* enc_w3 = (const float*)d_in[5];
  const float* enc_r1a = (const float*)d_in[6];
  const float* enc_r1b = (const float*)d_in[7];
  const float* enc_r2a = (const float*)d_in[8];
  const float* enc_r2b = (const float*)d_in[9];
  const float* wq = (const float*)d_in[10];
  const float* wk = (const float*)d_in[11];
  const float* wv = (const float*)d_in[12];
  const float* wo = (const float*)d_in[13];
  const float* reb_r1a = (const float*)d_in[14];
  const float* reb_r1b = (const float*)d_in[15];
  const float* reb_r2a = (const float*)d_in[16];
  const float* reb_r2b = (const float*)d_in[17];
  const float* cb = (const float*)d_in[18];
  const float* dec_w1 = (const float*)d_in[19];
  const float* dec_r1a = (const float*)d_in[20];
  const float* dec_r1b = (const float*)d_in[21];
  const float* dec_r2a = (const float*)d_in[22];
  const float* dec_r2b = (const float*)d_in[23];
  const float* dec_t1 = (const float*)d_in[24];
  const float* dec_t2 = (const float*)d_in[25];

  char* ws = (char*)d_ws;
  auto F = [&](size_t off) { return (float*)(ws + off); };

  // scratch layout (bytes)
  float* h1   = F(0);                      // [32,128,128,128] 268.4MB
  float* h2   = F(268435456);              // [32,128,64,64]   67.1MB
  float* h3   = F(335544320);              // [32,64,64,64]    33.6MB
  float* tA   = F(369098752);              // [32,64,64,64]
  float* h4   = F(0);                      // reuse (h1 dead)
  float* sEnc = F(33554432);
  float* kbuf = F(402653184);
  float* vbuf = F(403177472);
  float* esq  = F(403701760);
  unsigned int* counts = (unsigned int*)(ws + 403703808);
  int* idx    = (int*)(ws + 403707904);
  float* sr   = F(335544320);              // h3 slot
  float* sr2  = F(0);
  float* sr3  = F(33554432);
  float* quant = F(335544320);
  float* hd0  = F(0);                      // [32,128,64,64] 67MB
  float* td1  = F(67108864);
  float* hd1  = F(134217728);
  float* td2  = F(201326592);
  float* hd2  = F(268435456);
  float* t1o  = F(0);                      // [32,64,128,128] 134MB
  float* outr = (float*)d_out;

  // ---- encoder ----
  conv_wmma_kernel<<<dim3(256, 2, 32), 128, 0, stream>>>(imgh, enc_w1, nullptr, h1,
      3, 256, 256, 128, 128, 128, 4, 4, 2, 1, 0, 1);
  conv_wmma_kernel<<<dim3(64, 2, 32), 128, 0, stream>>>(h1, enc_w2, nullptr, h2,
      128, 128, 128, 128, 64, 64, 4, 4, 2, 1, 0, 1);
  conv_wmma_kernel<<<dim3(64, 1, 32), 128, 0, stream>>>(h2, enc_w3, nullptr, h3,
      128, 64, 64, 64, 64, 64, 3, 3, 1, 1, 0, 0);
  conv_wmma_kernel<<<dim3(64, 1, 32), 128, 0, stream>>>(h3, enc_r1a, nullptr, tA,
      64, 64, 64, 64, 64, 64, 3, 3, 1, 1, 1, 0);
  conv_wmma_kernel<<<dim3(64, 1, 32), 128, 0, stream>>>(tA, enc_r1b, h3, h4,
      64, 64, 64, 64, 64, 64, 1, 1, 1, 0, 1, 0);
  conv_wmma_kernel<<<dim3(64, 1, 32), 128, 0, stream>>>(h4, enc_r2a, nullptr, tA,
      64, 64, 64, 64, 64, 64, 3, 3, 1, 1, 1, 0);
  conv_wmma_kernel<<<dim3(64, 1, 32), 128, 0, stream>>>(tA, enc_r2b, h4, sEnc,
      64, 64, 64, 64, 64, 64, 1, 1, 1, 0, 1, 0);

  // ---- cross attention ----
  attn_kv_kernel<<<512, 256, 0, stream>>>(texth, wk, wv, kbuf, vbuf);
  attn_fused_kernel<<<32768, 128, 0, stream>>>(sEnc, kbuf, vbuf, wq, wo, mask, sr);

  // ---- rebuild resblocks ----
  conv_wmma_kernel<<<dim3(64, 1, 32), 128, 0, stream>>>(sr, reb_r1a, nullptr, tA,
      64, 64, 64, 64, 64, 64, 3, 3, 1, 1, 1, 0);
  conv_wmma_kernel<<<dim3(64, 1, 32), 128, 0, stream>>>(tA, reb_r1b, sr, sr2,
      64, 64, 64, 64, 64, 64, 1, 1, 1, 0, 1, 0);
  conv_wmma_kernel<<<dim3(64, 1, 32), 128, 0, stream>>>(sr2, reb_r2a, nullptr, tA,
      64, 64, 64, 64, 64, 64, 3, 3, 1, 1, 1, 0);
  conv_wmma_kernel<<<dim3(64, 1, 32), 128, 0, stream>>>(tA, reb_r2b, sr2, sr3,
      64, 64, 64, 64, 64, 64, 1, 1, 1, 0, 1, 0);

  // ---- vector quantization ----
  codebook_sq_kernel<<<2, 256, 0, stream>>>(cb, esq);
  zero_counts_kernel<<<2, 256, 0, stream>>>(counts);
  vq_argmin_kernel<<<8192, 32, 0, stream>>>(sr3, cb, esq, idx, counts);
  vq_gather_kernel<<<32768, 256, 0, stream>>>(idx, cb, quant);

  // ---- decoder ----
  conv_wmma_kernel<<<dim3(64, 2, 32), 128, 0, stream>>>(quant, dec_w1, nullptr, hd0,
      64, 64, 64, 128, 64, 64, 3, 3, 1, 1, 0, 0);
  conv_wmma_kernel<<<dim3(64, 2, 32), 128, 0, stream>>>(hd0, dec_r1a, nullptr, td1,
      128, 64, 64, 128, 64, 64, 3, 3, 1, 1, 1, 0);
  conv_wmma_kernel<<<dim3(64, 2, 32), 128, 0, stream>>>(td1, dec_r1b, hd0, hd1,
      128, 64, 64, 128, 64, 64, 1, 1, 1, 0, 1, 0);
  conv_wmma_kernel<<<dim3(64, 2, 32), 128, 0, stream>>>(hd1, dec_r2a, nullptr, td2,
      128, 64, 64, 128, 64, 64, 3, 3, 1, 1, 1, 0);
  conv_wmma_kernel<<<dim3(64, 2, 32), 128, 0, stream>>>(td2, dec_r2b, hd1, hd2,
      128, 64, 64, 128, 64, 64, 1, 1, 1, 0, 1, 0);
  convt_wmma_kernel<<<dim3(1024, 4, 32), 32, 0, stream>>>(hd2, dec_t1, t1o,
      128, 64, 64, 64, 1);
  convt_wmma_kernel<<<dim3(4096, 1, 32), 32, 0, stream>>>(t1o, dec_t2, outr,
      64, 128, 128, 3, 0);

  // ---- perplexity ----
  perplexity_kernel<<<1, 256, 0, stream>>>(counts, outr + 6291456);

  (void)in_sizes; (void)n_in; (void)out_size; (void)ws_size;
}
